// DirectedMultiheadAttention_79413945303742
// MI455X (gfx1250) — matrix operand order repaired and checked by reading, not compiled, not run on hardware
//
#include <hip/hip_runtime.h>

// ---------------------------------------------------------------------------
// DirectedMultiheadAttention for MI455X (gfx1250, wave32, WMMA 16x16x32 f16)
//   B=16, T=512, C=1024, H=16, Dh=64, TOPK=64, MAX_POS=512
// ---------------------------------------------------------------------------

#define BB   16
#define TT   512
#define CC   1024
#define HH   16
#define DH   64
#define TOPK 64

typedef _Float16 v16h __attribute__((ext_vector_type(16)));
typedef _Float16 v8h  __attribute__((ext_vector_type(8)));
typedef float    v8f  __attribute__((ext_vector_type(8)));

union V16H { v16h v; _Float16 h[16]; };

__device__ __forceinline__ v8f wmma_f16(v16h a, v16h b, v8f c) {
  // D = A(16x32 f16) * B(32x16 f16) + C(16x16 f32)
  return __builtin_amdgcn_wmma_f32_16x16x32_f16(
      /*neg_a=*/false, a, /*neg_b=*/false, b,
      /*c_mod=*/(short)0, c, /*reuse_a=*/false, /*reuse_b=*/false);
}

// ---------------------------------------------------------------------------
// Kernel 0: one-shot f32 -> f16 conversion (x and the 4 weight matrices).
// Runs once; ~12.6M elements, trivially bandwidth bound.  Removes all
// v_cvt_pk_f16_f32 from the hot GEMM loops and halves their load bytes.
// ---------------------------------------------------------------------------
__global__ __launch_bounds__(256) void cvt_f16_kernel(
    const float* __restrict__ src, _Float16* __restrict__ dst, int n4)
{
  const int i = blockIdx.x * 256 + threadIdx.x;
  if (i < n4) {
    const float4 v = ((const float4*)src)[i];
    _Float16* d = dst + (size_t)i * 4;
    d[0] = (_Float16)v.x; d[1] = (_Float16)v.y;
    d[2] = (_Float16)v.z; d[3] = (_Float16)v.w;
  }
}

// ---------------------------------------------------------------------------
// Kernel 1: fused QKV projection.  Y = x @ W.T + b  (8192 x 1024 x 1024, x3)
//   grid = (256 Mtiles of 32, 4 Ngroups, 3 matrices), block = 128 (4 waves)
//   each wave computes a 32x64 output tile: 2 A-frags share 4 B-frags
//   -> 6 fragment loads per 8 WMMAs (all pure-f16 b128 loads)
//   Q,K stored f16 [B,H,T,Dh];  V stored f16 transposed [B,H,Dh,T]
// ---------------------------------------------------------------------------
__global__ __launch_bounds__(128) void qkv_proj_kernel(
    const _Float16* __restrict__ xh,
    const _Float16* __restrict__ Wq16, const _Float16* __restrict__ Wk16,
    const _Float16* __restrict__ Wv16,
    const float* __restrict__ bq, const float* __restrict__ bk,
    const float* __restrict__ bv,
    _Float16* __restrict__ qh, _Float16* __restrict__ kh,
    _Float16* __restrict__ vTh)
{
  const int lane = threadIdx.x & 31;
  const int wave = threadIdx.x >> 5;
  const int m0 = blockIdx.x * 32;               // row tile (B*T rows)
  const int n0 = blockIdx.y * 256 + wave * 64;  // col base for this wave
  const int which = blockIdx.z;                 // 0=Q 1=K 2=V

  const _Float16* W  = (which == 0) ? Wq16 : (which == 1) ? Wk16 : Wv16;
  const float* bias  = (which == 0) ? bq   : (which == 1) ? bk   : bv;

  const int aRow = lane & 15;          // A: lane -> matrix row M
  const int aOff = (lane >> 4) << 3;   // A: +8 k-offset for upper half-wave
  const int bCol = lane & 15;          // B: lane -> matrix col N
  const int bOff = (lane >> 4) << 3;   // B: +8 k-offset for upper half-wave

  v8f acc[2][4] = {};
  const _Float16* arow0 = xh + (size_t)(m0 + aRow) * CC;
  const _Float16* arow1 = arow0 + (size_t)16 * CC;

  for (int kt = 0; kt < CC / 32; ++kt) {
    const int kb = kt * 32;
    __builtin_prefetch(arow0 + kb + 32, 0, 1);
    V16H a0, a1;
    *(v8h*)&a0.h[0] = *(const v8h*)(arow0 + kb + aOff);
    *(v8h*)&a0.h[8] = *(const v8h*)(arow0 + kb + aOff + 16);
    *(v8h*)&a1.h[0] = *(const v8h*)(arow1 + kb + aOff);
    *(v8h*)&a1.h[8] = *(const v8h*)(arow1 + kb + aOff + 16);
#pragma unroll
    for (int c = 0; c < 4; ++c) {
      const _Float16* bp = W + (size_t)(n0 + c * 16 + bCol) * CC + kb + bOff;
      V16H bf;
      *(v8h*)&bf.h[0] = *(const v8h*)(bp);
      *(v8h*)&bf.h[8] = *(const v8h*)(bp + 16);
      acc[0][c] = wmma_f16(a0.v, bf.v, acc[0][c]);
      acc[1][c] = wmma_f16(a1.v, bf.v, acc[1][c]);
    }
  }

  // Epilogue: +bias, scatter f16 into attention layouts
#pragma unroll
  for (int u = 0; u < 2; ++u) {
#pragma unroll
    for (int c = 0; c < 4; ++c) {
      const int col = n0 + c * 16 + (lane & 15);  // 0..1023
      const float bv_ = bias[col];
      const int h = col >> 6;   // head
      const int d = col & 63;   // dim within head
#pragma unroll
      for (int r = 0; r < 8; ++r) {
        const int m = m0 + u * 16 + r + ((lane >> 4) << 3);  // global row
        const int b = m >> 9;
        const int t = m & 511;
        const float y = acc[u][c][r] + bv_;
        if (which == 2) {
          vTh[(((size_t)(b * HH + h) * DH + d) * TT) + t] = (_Float16)y;
        } else {
          _Float16* dst = (which == 0) ? qh : kh;
          dst[(((size_t)(b * HH + h) * TT + t) * DH) + d] = (_Float16)y;
        }
      }
    }
  }
}

// ---------------------------------------------------------------------------
// Kernel 2: attention core.  One block per (b, h, 16-query-row tile).
//   scores (WMMA) -> LDS -> exact top-64 radix select -> softmax*gate -> P@V
//   grid = B*H*(T/16) = 8192, block = 128 (4 waves)
// ---------------------------------------------------------------------------
#define LDS_STRIDE 516  // 516 % 64 = 4 banks -> conflict-free fragment gathers

__global__ __launch_bounds__(128) void attn_kernel(
    const _Float16* __restrict__ qh,   // [B,H,T,Dh]
    const _Float16* __restrict__ kh,   // [B,H,T,Dh]
    const _Float16* __restrict__ vTh,  // [B,H,Dh,T]
    const float* __restrict__ rpb,     // [2*512-1, H]
    const float* __restrict__ gates,   // [H]
    _Float16* __restrict__ aoh)        // [B,T,H,Dh]
{
  __shared__ float sc[16 * LDS_STRIDE];

  const int lane = threadIdx.x & 31;
  const int wave = threadIdx.x >> 5;
  const int blk   = blockIdx.x;
  const int ttile = blk & 31;       // T/16 = 32 tiles
  const int bh    = blk >> 5;       // 0..255
  const int b = bh >> 4, h = bh & 15;
  const int t0 = ttile * 16;

  const _Float16* qbase = qh  + (size_t)bh * TT * DH;
  const _Float16* kbase = kh  + (size_t)bh * TT * DH;
  const _Float16* vbase = vTh + (size_t)bh * DH * TT;

  const int aRow = lane & 15;
  const int aOff = (lane >> 4) << 3;
  const int bOff = (lane >> 4) << 3;

  // ---- 1) scores = (Q K^T)/8 + rel_pos_bias, into LDS --------------------
  V16H a0, a1;  // Q fragments for k-steps d=[0,32) and d=[32,64)
  {
    const _Float16* ap = qbase + (size_t)(t0 + aRow) * DH + aOff;
    *(v8h*)&a0.h[0] = *(const v8h*)(ap);
    *(v8h*)&a0.h[8] = *(const v8h*)(ap + 16);
    *(v8h*)&a1.h[0] = *(const v8h*)(ap + 32);
    *(v8h*)&a1.h[8] = *(const v8h*)(ap + 48);
  }
  const float scale = 0.125f;  // 1/sqrt(64)

  for (int sti = 0; sti < 8; ++sti) {
    const int st = sti * 4 + wave;     // s-tile handled by this wave
    const int s0 = st * 16;
    v8f acc = {};
#pragma unroll
    for (int kt = 0; kt < 2; ++kt) {
      const _Float16* bp = kbase + (size_t)(s0 + (lane & 15)) * DH + kt * 32 + bOff;
      V16H bf;
      *(v8h*)&bf.h[0] = *(const v8h*)(bp);
      *(v8h*)&bf.h[8] = *(const v8h*)(bp + 16);
      acc = wmma_f16((kt == 0) ? a0.v : a1.v, bf.v, acc);
    }
    const int s = s0 + (lane & 15);
#pragma unroll
    for (int r = 0; r < 8; ++r) {
      const int trow = r + ((lane >> 4) << 3);
      const int t = t0 + trow;
      const float bias = rpb[(size_t)(s - t + 511) * HH + h];
      sc[trow * LDS_STRIDE + s] = acc[r] * scale + bias;
    }
  }
  __syncthreads();

  // ---- 2) per-row exact top-64 threshold + softmax + gate ----------------
  const float gate = gates[h];
  for (int rr = 0; rr < 4; ++rr) {
    const int row = wave * 4 + rr;
    float vals[16];
    unsigned keys[16];
#pragma unroll
    for (int j = 0; j < 16; ++j) {
      const float f = sc[row * LDS_STRIDE + j * 32 + lane];
      vals[j] = f;
      const unsigned u = __float_as_uint(f);
      // monotone map: order of keys == order of floats
      keys[j] = u ^ (((int)u >> 31) | 0x80000000u);
    }
    // radix select: find key of the 64th-largest element (exact, tie == ref)
    unsigned prefix = 0u;
    int kk = TOPK;
    for (int bit = 31; bit >= 0; --bit) {
      const unsigned trial = prefix | (1u << bit);
      int cnt = 0;
#pragma unroll
      for (int j = 0; j < 16; ++j) cnt += ((keys[j] >> bit) == (trial >> bit));
#pragma unroll
      for (int off = 16; off > 0; off >>= 1) cnt += __shfl_xor(cnt, off, 32);
      if (cnt >= kk) prefix = trial; else kk -= cnt;
    }
    // softmax over kept elements (keys >= prefix  <=>  scores >= kth)
    float rmax = -3.0e38f;
#pragma unroll
    for (int j = 0; j < 16; ++j)
      if (keys[j] >= prefix) rmax = fmaxf(rmax, vals[j]);
#pragma unroll
    for (int off = 16; off > 0; off >>= 1)
      rmax = fmaxf(rmax, __shfl_xor(rmax, off, 32));
    float rsum = 0.0f;
#pragma unroll
    for (int j = 0; j < 16; ++j) {
      const float e = (keys[j] >= prefix) ? __expf(vals[j] - rmax) : 0.0f;
      vals[j] = e;
      rsum += e;
    }
#pragma unroll
    for (int off = 16; off > 0; off >>= 1) rsum += __shfl_xor(rsum, off, 32);
    const float inv = gate / rsum;
#pragma unroll
    for (int j = 0; j < 16; ++j)
      sc[row * LDS_STRIDE + j * 32 + lane] = vals[j] * inv;
  }
  __syncthreads();

  // ---- 3) out = P @ V  (wave w -> d-tile w, 16 cols) ---------------------
  v8f acc = {};
  const int dcol = wave * 16 + (lane & 15);
  for (int st = 0; st < 16; ++st) {
    const float* pp = &sc[aRow * LDS_STRIDE + st * 32 + aOff];
    V16H a;
#pragma unroll
    for (int e = 0; e < 8; ++e) {
      a.h[e]     = (_Float16)pp[e];
      a.h[e + 8] = (_Float16)pp[e + 16];
    }
    const _Float16* bp = vbase + (size_t)dcol * TT + st * 32 + bOff;
    V16H bf;
    *(v8h*)&bf.h[0] = *(const v8h*)(bp);
    *(v8h*)&bf.h[8] = *(const v8h*)(bp + 16);
    acc = wmma_f16(a.v, bf.v, acc);
  }
#pragma unroll
  for (int r = 0; r < 8; ++r) {
    const int t = t0 + r + ((lane >> 4) << 3);
    aoh[(((size_t)(b * TT + t) * HH + h) * DH) + dcol] = (_Float16)acc[r];
  }
}

// ---------------------------------------------------------------------------
// Kernel 3: output projection.  out = AO @ Wo.T + bo  (f32 out)
//   grid = (256 Mtiles of 32, 4 Ngroups), block = 128 (4 waves, 32x64/wave)
// ---------------------------------------------------------------------------
__global__ __launch_bounds__(128) void out_proj_kernel(
    const _Float16* __restrict__ ao,    // [8192, 1024] f16
    const _Float16* __restrict__ Wo16,  // [1024, 1024] f16
    const float* __restrict__ bo,
    float* __restrict__ out)            // [8192, 1024] f32
{
  const int lane = threadIdx.x & 31;
  const int wave = threadIdx.x >> 5;
  const int m0 = blockIdx.x * 32;
  const int n0 = blockIdx.y * 256 + wave * 64;

  const int aRow = lane & 15;
  const int aOff = (lane >> 4) << 3;
  const int bCol = lane & 15;
  const int bOff = (lane >> 4) << 3;

  v8f acc[2][4] = {};
  const _Float16* arow0 = ao + (size_t)(m0 + aRow) * CC;
  const _Float16* arow1 = arow0 + (size_t)16 * CC;

  for (int kt = 0; kt < CC / 32; ++kt) {
    const int kb = kt * 32;
    __builtin_prefetch(arow0 + kb + 32, 0, 1);
    V16H a0, a1;
    *(v8h*)&a0.h[0] = *(const v8h*)(arow0 + kb + aOff);
    *(v8h*)&a0.h[8] = *(const v8h*)(arow0 + kb + aOff + 16);
    *(v8h*)&a1.h[0] = *(const v8h*)(arow1 + kb + aOff);
    *(v8h*)&a1.h[8] = *(const v8h*)(arow1 + kb + aOff + 16);
#pragma unroll
    for (int c = 0; c < 4; ++c) {
      const _Float16* bp = Wo16 + (size_t)(n0 + c * 16 + bCol) * CC + kb + bOff;
      V16H bf;
      *(v8h*)&bf.h[0] = *(const v8h*)(bp);
      *(v8h*)&bf.h[8] = *(const v8h*)(bp + 16);
      acc[0][c] = wmma_f16(a0.v, bf.v, acc[0][c]);
      acc[1][c] = wmma_f16(a1.v, bf.v, acc[1][c]);
    }
  }
#pragma unroll
  for (int u = 0; u < 2; ++u) {
#pragma unroll
    for (int c = 0; c < 4; ++c) {
      const int col = n0 + c * 16 + (lane & 15);
      const float bv_ = bo[col];
#pragma unroll
      for (int r = 0; r < 8; ++r) {
        const int m = m0 + u * 16 + r + ((lane >> 4) << 3);
        out[(size_t)m * CC + col] = acc[u][c][r] + bv_;
      }
    }
  }
}

// ---------------------------------------------------------------------------
extern "C" void kernel_launch(void* const* d_in, const int* in_sizes, int n_in,
                              void* d_out, int out_size, void* d_ws, size_t ws_size,
                              hipStream_t stream) {
  (void)in_sizes; (void)n_in; (void)out_size; (void)ws_size;

  const float* x   = (const float*)d_in[0];
  const float* Wq  = (const float*)d_in[1];
  const float* bq  = (const float*)d_in[2];
  const float* Wk  = (const float*)d_in[3];
  const float* bk  = (const float*)d_in[4];
  const float* Wv  = (const float*)d_in[5];
  const float* bv  = (const float*)d_in[6];
  const float* Wo  = (const float*)d_in[7];
  const float* bo  = (const float*)d_in[8];
  const float* hg  = (const float*)d_in[9];
  const float* rpb = (const float*)d_in[10];
  float* out = (float*)d_out;

  // workspace: 5 x 16MB activations (f16) + 4 x 2MB weights (f16) = 88 MB
  // (everything stays resident in MI455X's 192 MB L2)
  const size_t NELT = (size_t)BB * TT * CC;  // 8,388,608
  const size_t WELT = (size_t)CC * CC;       // 1,048,576
  _Float16* qh   = (_Float16*)d_ws;
  _Float16* kh   = qh   + NELT;
  _Float16* vTh  = kh   + NELT;
  _Float16* aoh  = vTh  + NELT;
  _Float16* xh   = aoh  + NELT;
  _Float16* wq16 = xh   + NELT;
  _Float16* wk16 = wq16 + WELT;
  _Float16* wv16 = wk16 + WELT;
  _Float16* wo16 = wv16 + WELT;

  // one-shot precision conversion
  cvt_f16_kernel<<<dim3((int)(NELT / 4 / 256)), 256, 0, stream>>>(x,  xh,   (int)(NELT / 4));
  cvt_f16_kernel<<<dim3((int)(WELT / 4 / 256)), 256, 0, stream>>>(Wq, wq16, (int)(WELT / 4));
  cvt_f16_kernel<<<dim3((int)(WELT / 4 / 256)), 256, 0, stream>>>(Wk, wk16, (int)(WELT / 4));
  cvt_f16_kernel<<<dim3((int)(WELT / 4 / 256)), 256, 0, stream>>>(Wv, wv16, (int)(WELT / 4));
  cvt_f16_kernel<<<dim3((int)(WELT / 4 / 256)), 256, 0, stream>>>(Wo, wo16, (int)(WELT / 4));

  qkv_proj_kernel<<<dim3(256, 4, 3), 128, 0, stream>>>(
      xh, wq16, wk16, wv16, bq, bk, bv, qh, kh, vTh);
  attn_kernel<<<dim3(8192), 128, 0, stream>>>(qh, kh, vTh, rpb, hg, aoh);
  out_proj_kernel<<<dim3(256, 4), 128, 0, stream>>>(aoh, wo16, bo, out);
}
